// MultiHeadAttention_11682311045770
// MI455X (gfx1250) — matrix operand order repaired and checked by reading, not compiled, not run on hardware
//
#include <hip/hip_runtime.h>
#include <cstdint>
#include <cstddef>

// ---------------------------------------------------------------------------
// MI455X (gfx1250) MHA block.
// bf16 WMMA (v_wmma_f32_16x16x32_bf16) for all matmuls, flash attention,
// DPP16 butterfly softmax reductions, and TDM (tensor_load_to_lds) double-
// buffered tile staging with s_wait_tensorcnt when the toolchain exposes the
// builtin (manual global->LDS staging fallback otherwise). Wave32 throughout.
// ---------------------------------------------------------------------------

typedef __bf16 bf16;
typedef bf16  v4bf  __attribute__((ext_vector_type(4)));
typedef bf16  v8bf  __attribute__((ext_vector_type(8)));
typedef bf16  v16bf __attribute__((ext_vector_type(16)));
typedef float v8f   __attribute__((ext_vector_type(8)));

#define B_DIM  4
#define NSEQ   2048
#define DM     1024       // D_MODEL == IN_DIM == 1024
#define H_DIM  16
#define HD     64
#define MROWS  (B_DIM*NSEQ)   // 8192
#define KDIM   1024
#define NDIM   1024

static constexpr int OUT_F32       = 0;
static constexpr int OUT_BF16      = 1;
static constexpr int OUT_BF16_BHND = 2;

// -------- Tensor Data Mover availability (probe both toolchains) -----------
#if defined(__HIP_DEVICE_COMPILE__) && \
    __has_builtin(__builtin_amdgcn_tensor_load_to_lds) && \
    __has_builtin(__builtin_amdgcn_s_wait_tensorcnt)
#define HAVE_TDM 1
#else
#define HAVE_TDM 0
#endif

#if HAVE_TDM
typedef unsigned int v4u __attribute__((ext_vector_type(4)));
typedef int          v4i __attribute__((ext_vector_type(4)));
typedef int          v8i __attribute__((ext_vector_type(8)));

// Low 32 bits of a generic (__shared__) address == wave-relative LDS offset.
__device__ __forceinline__ unsigned lds_off(const void* p) {
  return (unsigned)(size_t)p;
}

// 2D bf16 tile DMA: global (row stride stride0 elems) -> LDS, optional LDS
// row padding (padIv/padAmt are the encoded D# codes). ISA 08_async_tensor §8.
__device__ __forceinline__ void tdm_load_2d_bf16(const bf16* gsrc, unsigned ldsoff,
                                                 unsigned dim0, unsigned dim1,
                                                 unsigned stride0,
                                                 unsigned tile0, unsigned tile1,
                                                 bool pad, unsigned padIvCode,
                                                 unsigned padAmtCode) {
  unsigned long long ga = (unsigned long long)(size_t)gsrc;   // byte address
  v4u g0;
  g0[0] = 1u;                                        // count=1, user mode
  g0[1] = ldsoff;                                    // lds_addr
  g0[2] = (unsigned)ga;                              // global_addr[31:0]
  g0[3] = ((unsigned)(ga >> 32) & 0x01FFFFFFu) | (2u << 30);  // [56:32]|type=2
  v8i g1;
  unsigned w0 = (1u << 16);                          // data_size = 2 bytes
  if (pad) w0 |= (1u << 20) | (padIvCode << 22) | (padAmtCode << 25);
  g1[0] = (int)w0;
  g1[1] = (int)((dim0 & 0xFFFFu) << 16);             // tensor_dim0[15:0]
  g1[2] = (int)((dim0 >> 16) | ((dim1 & 0xFFFFu) << 16));
  g1[3] = (int)((dim1 >> 16) | (tile0 << 16));       // tile_dim0
  g1[4] = (int)(tile1 & 0xFFFFu);                    // tile_dim1 (tile_dim2=0)
  g1[5] = (int)stride0;                              // tensor_dim0_stride[31:0]
  g1[6] = 0;
  g1[7] = 0;
  v4i z4 = {0, 0, 0, 0};
#if defined(__clang_major__) && (__clang_major__ >= 23)
  v8i z8 = {0, 0, 0, 0, 0, 0, 0, 0};
  __builtin_amdgcn_tensor_load_to_lds(g0, g1, z4, z4, z8, 0);
#else
  __builtin_amdgcn_tensor_load_to_lds(g0, g1, z4, z4, 0);
#endif
}
#endif  // HAVE_TDM

__device__ __forceinline__ v16bf pack16(v8bf lo, v8bf hi) {
  v16bf r;
#pragma unroll
  for (int i = 0; i < 8; i++) { r[i] = lo[i]; r[8 + i] = hi[i]; }
  return r;
}

// ---- DPP16 cross-lane helpers (wave32; reduction within 16-lane group) ----
template <int CTRL>
__device__ __forceinline__ float dppf(float x) {
  int i = __builtin_bit_cast(int, x);
  i = __builtin_amdgcn_update_dpp(0, i, CTRL, 0xF, 0xF, true);
  return __builtin_bit_cast(float, i);
}
__device__ __forceinline__ float red16_max(float x) {
  x = fmaxf(x, dppf<0xB1>(x));    // quad_perm(1,0,3,2)
  x = fmaxf(x, dppf<0x4E>(x));    // quad_perm(2,3,0,1)
  x = fmaxf(x, dppf<0x141>(x));   // row_half_mirror
  x = fmaxf(x, dppf<0x140>(x));   // row_mirror
  return x;
}
__device__ __forceinline__ float red16_sum(float x) {
  x += dppf<0xB1>(x);
  x += dppf<0x4E>(x);
  x += dppf<0x141>(x);
  x += dppf<0x140>(x);
  return x;
}

// ---------------- f32 -> bf16 elementwise convert --------------------------
__global__ __launch_bounds__(256) void cvt_f32_bf16(const float* __restrict__ in,
                                                    bf16* __restrict__ out, int n) {
  int i = (blockIdx.x * 256 + threadIdx.x) * 4;
  if (i + 3 < n) {
    float4 v = *(const float4*)(in + i);
    v4bf o;
    o[0] = (bf16)v.x; o[1] = (bf16)v.y; o[2] = (bf16)v.z; o[3] = (bf16)v.w;
    *(v4bf*)(out + i) = o;
  }
}

// ------------- weight f32 [K][N] -> bf16 transposed [N][K] -----------------
__global__ void wtrans_f32_bf16t(const float* __restrict__ W, bf16* __restrict__ Wt) {
  __shared__ float tile[32][33];
  int n0 = blockIdx.x * 32, k0 = blockIdx.y * 32;
  int tx = threadIdx.x, ty = threadIdx.y;
#pragma unroll
  for (int i = 0; i < 4; i++)
    tile[ty + i * 8][tx] = W[(size_t)(k0 + ty + i * 8) * NDIM + n0 + tx];
  __syncthreads();
#pragma unroll
  for (int i = 0; i < 4; i++)
    Wt[(size_t)(n0 + ty + i * 8) * KDIM + k0 + tx] = (bf16)tile[tx][ty + i * 8];
}

// ------------------------- bf16 WMMA GEMM ----------------------------------
// C[M=8192][N=1024] = A[M][K=1024] * Bt[N][K]^T + bias, templated epilogue.
// Block tile 128x128, 8 waves (4x2), 32x64 per wave, K stage 64.
// TDM path: ping-pong LDS, DMA of stage s+1 overlapped with WMMA on stage s.
template <int MODE>
__global__ __launch_bounds__(256) void gemm_bf16(const bf16* __restrict__ A,
                                                 const bf16* __restrict__ Bt,
                                                 const float* __restrict__ bias,
                                                 void* __restrict__ outp) {
#if HAVE_TDM
  __shared__ bf16 As[2][128][72];   // 64 + 8 pad (TDM pad: 4 DW every 32 DW)
  __shared__ bf16 Bs[2][128][72];
#else
  __shared__ bf16 As[1][128][72];
  __shared__ bf16 Bs[1][128][72];
#endif
  const int tid  = threadIdx.x;
  const int wid  = tid >> 5, lane = tid & 31;
  const int lr   = lane & 15, lhi = lane >> 4;
  const int wm   = wid >> 1, wn = wid & 1;      // 4x2 wave grid, 32x64 per wave
  const int m0   = blockIdx.y * 128;
  const int n0   = blockIdx.x * 128;

  const v8f vzero = {0.f, 0.f, 0.f, 0.f, 0.f, 0.f, 0.f, 0.f};
  v8f acc[2][4];
#pragma unroll
  for (int mi = 0; mi < 2; mi++)
#pragma unroll
    for (int ni = 0; ni < 4; ni++) acc[mi][ni] = vzero;

  auto compute_stage = [&](const bf16 (*Asb)[72], const bf16 (*Bsb)[72]) {
#pragma unroll
    for (int kh = 0; kh < 2; kh++) {    // two K=32 sub-steps per stage
      const int kc = kh * 32;
      v16bf af[2], bfr[4];
#pragma unroll
      for (int mi = 0; mi < 2; mi++) {  // A frag: chunks at K=off and K=off+16
        const bf16* p = &Asb[wm * 32 + mi * 16 + lr][kc + lhi * 8];
        af[mi] = pack16(*(const v8bf*)p, *(const v8bf*)(p + 16));
      }
#pragma unroll
      for (int ni = 0; ni < 4; ni++) {  // B frag: chunks at K=off and K=off+16
        const bf16* p = &Bsb[wn * 64 + ni * 16 + lr][kc + lhi * 8];
        bfr[ni] = pack16(*(const v8bf*)p, *(const v8bf*)(p + 16));
      }
#pragma unroll
      for (int mi = 0; mi < 2; mi++)
#pragma unroll
        for (int ni = 0; ni < 4; ni++)
          acc[mi][ni] = __builtin_amdgcn_wmma_f32_16x16x32_bf16(
              false, af[mi], false, bfr[ni], (short)0, acc[mi][ni], false, false);
    }
  };

#if HAVE_TDM
  if (wid == 0) {   // prologue: DMA stage 0, wait, release
    tdm_load_2d_bf16(A  + (size_t)m0 * KDIM, lds_off(&As[0][0][0]),
                     64, 128, KDIM, 64, 128, true, 4, 3);
    tdm_load_2d_bf16(Bt + (size_t)n0 * KDIM, lds_off(&Bs[0][0][0]),
                     64, 128, KDIM, 64, 128, true, 4, 3);
    __builtin_amdgcn_s_wait_tensorcnt((short)0);
  }
  __syncthreads();
  for (int s = 0; s < KDIM / 64; s++) {
    const int cur = s & 1;
    const int kkN = (s + 1) * 64;
    if (wid == 0 && kkN < KDIM) {       // DMA next stage into other buffer
      tdm_load_2d_bf16(A  + (size_t)m0 * KDIM + kkN, lds_off(&As[cur ^ 1][0][0]),
                       64, 128, KDIM, 64, 128, true, 4, 3);
      tdm_load_2d_bf16(Bt + (size_t)n0 * KDIM + kkN, lds_off(&Bs[cur ^ 1][0][0]),
                       64, 128, KDIM, 64, 128, true, 4, 3);
    }
    compute_stage(As[cur], Bs[cur]);    // WMMA overlapped with TDM DMA
    if (wid == 0) __builtin_amdgcn_s_wait_tensorcnt((short)0);
    __syncthreads();
  }
#else
  for (int kk = 0; kk < KDIM; kk += 64) {
    __syncthreads();
#pragma unroll
    for (int i = 0; i < 4; i++) {
      int c   = tid + i * 256;          // 1024 16B chunks per 128x64 tile
      int row = c >> 3;
      int co  = (c & 7) * 8;
      *(v8bf*)&As[0][row][co] = *(const v8bf*)(A  + (size_t)(m0 + row) * KDIM + kk + co);
      *(v8bf*)&Bs[0][row][co] = *(const v8bf*)(Bt + (size_t)(n0 + row) * KDIM + kk + co);
      if (kk + 64 < KDIM) {
        __builtin_prefetch(A  + (size_t)(m0 + row) * KDIM + kk + 64 + co, 0, 0);
        __builtin_prefetch(Bt + (size_t)(n0 + row) * KDIM + kk + 64 + co, 0, 0);
      }
    }
    __syncthreads();
    compute_stage(As[0], Bs[0]);
  }
#endif

#pragma unroll
  for (int mi = 0; mi < 2; mi++) {
#pragma unroll
    for (int ni = 0; ni < 4; ni++) {
      int ncol = n0 + wn * 64 + ni * 16 + lr;
      float bv = bias[ncol];
#pragma unroll
      for (int r = 0; r < 8; r++) {
        int m = m0 + wm * 32 + mi * 16 + r + lhi * 8;
        float val = acc[mi][ni][r] + bv;
        if (MODE == OUT_F32) {
          ((float*)outp)[(size_t)m * NDIM + ncol] = val;
        } else if (MODE == OUT_BF16) {
          ((bf16*)outp)[(size_t)m * NDIM + ncol] = (bf16)val;
        } else {  // [B,N,D] -> [B,H,N,HD]
          int b = m >> 11, n = m & (NSEQ - 1);
          int h = ncol >> 6, d = ncol & 63;
          ((bf16*)outp)[(((size_t)(b * H_DIM + h) * NSEQ + n) << 6) + d] = (bf16)val;
        }
      }
    }
  }
}

// --------------- RoPE + per-head RMSNorm (one wave per row) ----------------
__global__ __launch_bounds__(256) void rope_rms(const bf16* __restrict__ X,
                                                const float* __restrict__ w,
                                                bf16* __restrict__ out,
                                                float outScale) {
  int rid  = blockIdx.x * 8 + (threadIdx.x >> 5);  // (b,n,h) row id
  int lane = threadIdx.x & 31;
  int h  = rid & (H_DIM - 1);
  int bn = rid >> 4;                // b*NSEQ + n
  int n  = bn & (NSEQ - 1);
  int b  = bn >> 11;
  const bf16* src = X + (size_t)bn * DM + h * HD;
  float x1 = (float)src[lane];
  float x2 = (float)src[lane + 32];
  float inv = __expf(-(float)lane * (9.2103403719761836f / 32.f));  // 10000^(-l/32)
  float ang = (float)n * inv;
  float s, c;
  __sincosf(ang, &s, &c);
  float o1 = x1 * c - x2 * s;
  float o2 = x2 * c + x1 * s;
  float ss = o1 * o1 + o2 * o2;
#pragma unroll
  for (int off = 16; off > 0; off >>= 1) ss += __shfl_xor(ss, off, 32);
  float r = rsqrtf(ss * (1.f / 64.f) + 1e-6f) * outScale;
  bf16* dst = out + ((size_t)(b * H_DIM + h) * NSEQ + n) * HD;
  dst[lane]      = (bf16)(o1 * r * w[lane]);
  dst[lane + 32] = (bf16)(o2 * r * w[lane + 32]);
}

// --------------------------- flash attention -------------------------------
// grid (NSEQ/128, B*H); 8 waves x 16 q-rows; key tiles of 64.
// TDM path: K tiles DMA'd ping-pong; V tiles double-buffered manually
// (transpose); one barrier per key tile.
__global__ __launch_bounds__(256) void attn_fa(const bf16* __restrict__ qh,
                                               const bf16* __restrict__ kh,
                                               const bf16* __restrict__ vh,
                                               bf16* __restrict__ out) {
#if HAVE_TDM
  __shared__ bf16 Ks[2][64][72];     // [key][d]
  __shared__ bf16 Vt[2][64][72];     // [d][key]  (transposed V)
#else
  __shared__ bf16 Ks[1][64][72];
  __shared__ bf16 Vt[1][64][72];
#endif
  __shared__ bf16 Ps[8][16][72];     // per-wave P scratch
  const int tid = threadIdx.x;
  const int wid = tid >> 5, lane = tid & 31;
  const int lr  = lane & 15, lhi = lane >> 4;
  const int bh  = blockIdx.y;
  const size_t base = (size_t)bh * NSEQ * HD;
  const int q0 = blockIdx.x * 128 + wid * 16;
  const int vkey = tid >> 2, vd0 = (tid & 3) * 16;   // V-transpose mapping

  v16bf qf[2];
  {
    const bf16* p = qh + base + (size_t)(q0 + lr) * HD + lhi * 8;
#pragma unroll
    for (int kk = 0; kk < 2; kk++)
      qf[kk] = pack16(*(const v8bf*)(p + kk * 32), *(const v8bf*)(p + kk * 32 + 16));
  }

  const v8f vzero = {0.f, 0.f, 0.f, 0.f, 0.f, 0.f, 0.f, 0.f};
  v8f acc_o[4];
#pragma unroll
  for (int d4 = 0; d4 < 4; d4++) acc_o[d4] = vzero;
  float mrow[8], lrow[8];
#pragma unroll
  for (int r = 0; r < 8; r++) { mrow[r] = -3.0e38f; lrow[r] = 0.f; }

  // per-tile compute pieces -------------------------------------------------
  auto s_softmax_p = [&](const bf16 (*KsC)[72], v16bf* pf) {
    v8f sv[4];
#pragma unroll
    for (int n4 = 0; n4 < 4; n4++) {
      v8f s = vzero;
#pragma unroll
      for (int kk = 0; kk < 2; kk++) {
        const bf16* p = &KsC[n4 * 16 + lr][kk * 32 + lhi * 8];
        v16bf bfr = pack16(*(const v8bf*)p, *(const v8bf*)(p + 16));
        s = __builtin_amdgcn_wmma_f32_16x16x32_bf16(false, qf[kk], false, bfr,
                                                    (short)0, s, false, false);
      }
      sv[n4] = s;
    }
    float mt[8];
#pragma unroll
    for (int r = 0; r < 8; r++)
      mt[r] = red16_max(fmaxf(fmaxf(sv[0][r], sv[1][r]), fmaxf(sv[2][r], sv[3][r])));
    float corr[8], rs[8];
#pragma unroll
    for (int r = 0; r < 8; r++) {
      float mn = fmaxf(mrow[r], mt[r]);
      corr[r] = __expf(mrow[r] - mn);
      mrow[r] = mn;
      rs[r] = 0.f;
    }
#pragma unroll
    for (int n4 = 0; n4 < 4; n4++)
#pragma unroll
      for (int r = 0; r < 8; r++) {
        float pv = __expf(sv[n4][r] - mrow[r]);
        sv[n4][r] = pv;
        rs[r] += pv;
      }
#pragma unroll
    for (int r = 0; r < 8; r++) rs[r] = red16_sum(rs[r]);
#pragma unroll
    for (int r = 0; r < 8; r++) lrow[r] = lrow[r] * corr[r] + rs[r];
#pragma unroll
    for (int d4 = 0; d4 < 4; d4++)
#pragma unroll
      for (int r = 0; r < 8; r++) acc_o[d4][r] *= corr[r];
#pragma unroll
    for (int n4 = 0; n4 < 4; n4++)   // P (C layout) -> per-wave LDS
#pragma unroll
      for (int r = 0; r < 8; r++)
        Ps[wid][r + lhi * 8][n4 * 16 + lr] = (bf16)sv[n4][r];
#pragma unroll
    for (int kk = 0; kk < 2; kk++) { // reload as A fragments
      const bf16* p = &Ps[wid][lr][kk * 32 + lhi * 8];
      pf[kk] = pack16(*(const v8bf*)p, *(const v8bf*)(p + 16));
    }
  };
  auto pv_accum = [&](const bf16 (*VtC)[72], const v16bf* pf) {
#pragma unroll
    for (int d4 = 0; d4 < 4; d4++)
#pragma unroll
      for (int kk = 0; kk < 2; kk++) {
        const bf16* p = &VtC[d4 * 16 + lr][kk * 32 + lhi * 8];
        v16bf vfr = pack16(*(const v8bf*)p, *(const v8bf*)(p + 16));
        acc_o[d4] = __builtin_amdgcn_wmma_f32_16x16x32_bf16(
            false, pf[kk], false, vfr, (short)0, acc_o[d4], false, false);
      }
  };
  const int NT = NSEQ / 64;

#if HAVE_TDM
  // prologue: DMA K tile 0, manual-transpose V tile 0
  if (wid == 0)
    tdm_load_2d_bf16(kh + base, lds_off(&Ks[0][0][0]), 64, 64, HD, 64, 64, true, 4, 3);
  {
    const bf16* vp = vh + base + (size_t)vkey * HD + vd0;
    v8bf a = *(const v8bf*)vp, b2 = *(const v8bf*)(vp + 8);
#pragma unroll
    for (int j = 0; j < 8; j++) { Vt[0][vd0 + j][vkey] = a[j]; Vt[0][vd0 + 8 + j][vkey] = b2[j]; }
  }
  if (wid == 0) __builtin_amdgcn_s_wait_tensorcnt((short)0);
  __syncthreads();
  for (int kt = 0; kt < NT; ++kt) {
    const int cur = kt & 1;
    const bool haveNext = (kt + 1) < NT;
    if (wid == 0 && haveNext)            // DMA next K tile into other buffer
      tdm_load_2d_bf16(kh + base + (size_t)(kt + 1) * 64 * HD,
                       lds_off(&Ks[cur ^ 1][0][0]), 64, 64, HD, 64, 64, true, 4, 3);
    v8bf va = {}, vb2 = {};
    if (haveNext) {                      // issue next V tile loads early
      const bf16* vp = vh + base + (size_t)((kt + 1) * 64 + vkey) * HD + vd0;
      va = *(const v8bf*)vp; vb2 = *(const v8bf*)(vp + 8);
    }
    v16bf pf[2];
    s_softmax_p(Ks[cur], pf);            // S = Q*K^T, online softmax, P frags
    if (haveNext) {                      // store next V (other buffer)
#pragma unroll
      for (int j = 0; j < 8; j++) {
        Vt[cur ^ 1][vd0 + j][vkey] = va[j];
        Vt[cur ^ 1][vd0 + 8 + j][vkey] = vb2[j];
      }
    }
    pv_accum(Vt[cur], pf);               // O += P*V
    if (wid == 0) __builtin_amdgcn_s_wait_tensorcnt((short)0);
    __syncthreads();
  }
#else
  for (int kt = 0; kt < NT; ++kt) {
    __syncthreads();
#pragma unroll
    for (int i = 0; i < 2; i++) {        // K tile, coalesced
      int c = tid + i * 256;
      int row = c >> 3, co = (c & 7) * 8;
      *(v8bf*)&Ks[0][row][co] =
          *(const v8bf*)(kh + base + (size_t)(kt * 64 + row) * HD + co);
    }
    {                                    // V tile, transposed
      const bf16* vp = vh + base + (size_t)(kt * 64 + vkey) * HD + vd0;
      v8bf a = *(const v8bf*)vp, b2 = *(const v8bf*)(vp + 8);
#pragma unroll
      for (int j = 0; j < 8; j++) { Vt[0][vd0 + j][vkey] = a[j]; Vt[0][vd0 + 8 + j][vkey] = b2[j]; }
    }
    __syncthreads();
    v16bf pf[2];
    s_softmax_p(Ks[0], pf);
    pv_accum(Vt[0], pf);
  }
#endif

  const int b = bh >> 4, h = bh & 15;
#pragma unroll
  for (int d4 = 0; d4 < 4; d4++)
#pragma unroll
    for (int r = 0; r < 8; r++) {
      float val = acc_o[d4][r] / lrow[r];
      int row = q0 + r + lhi * 8;
      int col = h * HD + d4 * 16 + lr;
      out[(size_t)(b * NSEQ + row) * DM + col] = (bf16)val;  // [B,N,D] layout
    }
}

// ---------------------------------------------------------------------------
extern "C" void kernel_launch(void* const* d_in, const int* in_sizes, int n_in,
                              void* d_out, int out_size, void* d_ws, size_t ws_size,
                              hipStream_t stream) {
  (void)in_sizes; (void)n_in; (void)out_size; (void)ws_size;
  const float* q    = (const float*)d_in[0];
  const float* k    = (const float*)d_in[1];
  const float* v    = (const float*)d_in[2];
  const float* wq_w = (const float*)d_in[3];
  const float* wq_b = (const float*)d_in[4];
  const float* wk_w = (const float*)d_in[5];
  const float* wk_b = (const float*)d_in[6];
  const float* wv_w = (const float*)d_in[7];
  const float* wv_b = (const float*)d_in[8];
  const float* wo_w = (const float*)d_in[9];
  const float* wo_b = (const float*)d_in[10];
  const float* qn_w = (const float*)d_in[11];
  const float* kn_w = (const float*)d_in[12];

  char* ws = (char*)d_ws;
  const size_t MB = 1ull << 20;
  bf16* qb  = (bf16*)(ws + 0 * MB);     // bf16 inputs      (16 MB each)
  bf16* kb  = (bf16*)(ws + 16 * MB);
  bf16* vb  = (bf16*)(ws + 32 * MB);
  bf16* wqt = (bf16*)(ws + 48 * MB);    // transposed bf16 weights (2 MB each)
  bf16* wkt = (bf16*)(ws + 50 * MB);
  bf16* wvt = (bf16*)(ws + 52 * MB);
  bf16* wot = (bf16*)(ws + 54 * MB);
  bf16* qp  = (bf16*)(ws + 56 * MB);    // q/k projections [B,N,D] (16 MB each)
  bf16* kp  = (bf16*)(ws + 72 * MB);
  bf16* qhd = (bf16*)(ws + 88 * MB);    // [B,H,N,HD] bf16 (16 MB each)
  bf16* khd = (bf16*)(ws + 104 * MB);
  bf16* vhd = (bf16*)(ws + 120 * MB);
  bf16* att = (bf16*)(ws + 136 * MB);   // attention out [B,N,D] bf16 (16 MB)

  const int nElem = MROWS * DM;         // 8388608
  cvt_f32_bf16<<<nElem / 1024, 256, 0, stream>>>(q, qb, nElem);
  cvt_f32_bf16<<<nElem / 1024, 256, 0, stream>>>(k, kb, nElem);
  cvt_f32_bf16<<<nElem / 1024, 256, 0, stream>>>(v, vb, nElem);

  dim3 wg(32, 32), wb(32, 8);
  wtrans_f32_bf16t<<<wg, wb, 0, stream>>>(wq_w, wqt);
  wtrans_f32_bf16t<<<wg, wb, 0, stream>>>(wk_w, wkt);
  wtrans_f32_bf16t<<<wg, wb, 0, stream>>>(wv_w, wvt);
  wtrans_f32_bf16t<<<wg, wb, 0, stream>>>(wo_w, wot);

  dim3 gg(NDIM / 128, MROWS / 128);     // (8, 64)
  gemm_bf16<OUT_BF16><<<gg, 256, 0, stream>>>(qb, wqt, wq_b, qp);
  gemm_bf16<OUT_BF16><<<gg, 256, 0, stream>>>(kb, wkt, wk_b, kp);
  gemm_bf16<OUT_BF16_BHND><<<gg, 256, 0, stream>>>(vb, wvt, wv_b, vhd);

  // rows = B*N*H = 131072, 8 rows (waves) per block; q gets HD^-0.5 folded in
  rope_rms<<<16384, 256, 0, stream>>>(qp, qn_w, qhd, 0.125f);
  rope_rms<<<16384, 256, 0, stream>>>(kp, kn_w, khd, 1.0f);

  attn_fa<<<dim3(NSEQ / 128, B_DIM * H_DIM), 256, 0, stream>>>(qhd, khd, vhd, att);

  gemm_bf16<OUT_F32><<<gg, 256, 0, stream>>>(att, wot, wo_b, d_out);
}
